// MoEFFN_89309549953086
// MI455X (gfx1250) — matrix operand order repaired and checked by reading, not compiled, not run on hardware
//
#include <hip/hip_runtime.h>
#include <hip/hip_bf16.h>
#include <math.h>

// Problem constants (from reference): B=4, T=2048, C=1024, E=8
#define NTOK 8192   // B*T
#define CDIM 1024
#define NEXP 8
#define ROWP 1032   // padded LDS row stride in bf16 elements (1024 + 8 pad)

typedef __attribute__((ext_vector_type(16))) __bf16 v16bf;
typedef __attribute__((ext_vector_type(8)))  float  v8f;

// ---------------- kernel 0: zero per-expert counters ----------------
__global__ void moe_init_kernel(int* counts) {
    if (threadIdx.x < NEXP) counts[threadIdx.x] = 0;
}

// ---------------- kernel 1: router (one wave32 per token) ----------------
__global__ void moe_router_kernel(const float* __restrict__ x,
                                  const float* __restrict__ Wr,
                                  const float* __restrict__ br,
                                  float* __restrict__ out,
                                  int* __restrict__ counts,
                                  int* __restrict__ tokenList) {
    const int lane   = threadIdx.x & 31;
    const int waveId = threadIdx.x >> 5;
    const int token  = blockIdx.x * 8 + waveId;   // 8 waves per block
    if (token >= NTOK) return;

    const float* xrow = x + (size_t)token * CDIM;

    float acc[NEXP];
#pragma unroll
    for (int e = 0; e < NEXP; ++e) acc[e] = 0.0f;

    for (int c = lane; c < CDIM; c += 32) {
        const float xv = xrow[c];
#pragma unroll
        for (int e = 0; e < NEXP; ++e)
            acc[e] = fmaf(xv, Wr[e * CDIM + c], acc[e]);
    }

    // wave32 butterfly reduction (warpSize == 32 on gfx1250)
#pragma unroll
    for (int e = 0; e < NEXP; ++e) {
#pragma unroll
        for (int off = 16; off > 0; off >>= 1)
            acc[e] += __shfl_xor(acc[e], off, 32);
        acc[e] += br[e];
    }

    float m = acc[0];
#pragma unroll
    for (int e = 1; e < NEXP; ++e) m = fmaxf(m, acc[e]);
    float ex[NEXP];
    float s = 0.0f;
#pragma unroll
    for (int e = 0; e < NEXP; ++e) { ex[e] = __expf(acc[e] - m); s += ex[e]; }

    int sel = -1;
#pragma unroll
    for (int e = 0; e < NEXP; ++e)
        if (ex[e] > 0.5f * s) sel = e;   // at most one can exceed 0.5

    if (sel >= 0) {
        if (lane == 0) {
            int pos = atomicAdd(&counts[sel], 1);
            tokenList[sel * NTOK + pos] = token;
        }
    } else {
        float4  z  = make_float4(0.f, 0.f, 0.f, 0.f);
        float4* o4 = (float4*)(out + (size_t)token * CDIM);
#pragma unroll
        for (int i = 0; i < 8; ++i) o4[lane + 32 * i] = z;
    }
}

// load 16 contiguous f32 and convert to a bf16 B-fragment register
__device__ __forceinline__ v16bf loadB_bf16(const float* __restrict__ pb) {
    float4 b0 = *(const float4*)(pb);
    float4 b1 = *(const float4*)(pb + 4);
    float4 b2 = *(const float4*)(pb + 8);
    float4 b3 = *(const float4*)(pb + 12);
    v16bf Bv;
    Bv[0]=(__bf16)b0.x;  Bv[1]=(__bf16)b0.y;  Bv[2]=(__bf16)b0.z;  Bv[3]=(__bf16)b0.w;
    Bv[4]=(__bf16)b1.x;  Bv[5]=(__bf16)b1.y;  Bv[6]=(__bf16)b1.z;  Bv[7]=(__bf16)b1.w;
    Bv[8]=(__bf16)b2.x;  Bv[9]=(__bf16)b2.y;  Bv[10]=(__bf16)b2.z; Bv[11]=(__bf16)b2.w;
    Bv[12]=(__bf16)b3.x; Bv[13]=(__bf16)b3.y; Bv[14]=(__bf16)b3.z; Bv[15]=(__bf16)b3.w;
    return Bv;
}

// ---------------- kernel 2: grouped expert GEMM via bf16 WMMA ----------------
// grid: (Mtiles=512, CDIM/256, NEXP). Block = 256 threads = 8 waves.
// A tile (16 tokens x 1024 K) staged once into LDS as bf16 (padded rows).
// Each wave owns TWO 16x16 output tiles (32 columns) sharing each A fragment:
// 2 WMMAs per A ds_load pair, independent accumulators -> no RAW chain.
__global__ void moe_expert_gemm_kernel(const float* __restrict__ x,
                                       const float* __restrict__ We,
                                       const float* __restrict__ be,
                                       float* __restrict__ out,
                                       const int* __restrict__ counts,
                                       const int* __restrict__ tokenList) {
    __shared__ unsigned short sA[16 * ROWP];   // bf16 bits, 16 rows, padded
    __shared__ int sTok[16];

    const int e     = blockIdx.z;
    const int count = counts[e];
    const int m0    = blockIdx.x * 16;
    if (m0 >= count) return;              // block-uniform exit: EXEC all-1s

    const int lane   = threadIdx.x & 31;
    const int waveId = threadIdx.x >> 5;
    const int r  = lane & 15;
    const int hi = lane >> 4;

    // stage gathered token indices (clamped for ragged tail)
    if (threadIdx.x < 16) {
        int gm = m0 + threadIdx.x;
        sTok[threadIdx.x] = tokenList[e * NTOK + (gm < count ? gm : count - 1)];
    }
    __syncthreads();

    // ---- cooperative A-tile fill: 16 rows x 1024 f32 -> bf16 in LDS ----
#pragma unroll
    for (int i = 0; i < 16; ++i) {
        const int idx = threadIdx.x + 256 * i;
        const int row = idx >> 8;          // 256 float4 per row
        const int c4  = idx & 255;
        float4 f = *(const float4*)(x + (size_t)sTok[row] * CDIM + c4 * 4);
        union { __bf16 h[4]; uint2 u; } p;
        p.h[0] = (__bf16)f.x; p.h[1] = (__bf16)f.y;
        p.h[2] = (__bf16)f.z; p.h[3] = (__bf16)f.w;
        *(uint2*)&sA[row * ROWP + c4 * 4] = p.u;
    }
    __syncthreads();

    // ---- K loop: A from LDS (shared by 2 N-tiles), B from global ----
    const int n0 = blockIdx.y * 256 + waveId * 32 + r;   // first column tile
    const int n1 = n0 + 16;                              // second column tile
    const float* wrow0 = We + ((size_t)e * CDIM + n0) * CDIM;
    const float* wrow1 = We + ((size_t)e * CDIM + n1) * CDIM;
    const unsigned short* aRow = &sA[r * ROWP];

    v8f acc0 = {};
    v8f acc1 = {};

#pragma unroll 2
    for (int kk = 0; kk < CDIM; kk += 32) {
        // A fragment: lanes 0-15: K = kk+0..15; lanes 16-31: K = kk+16..31
        union { v16bf v; uint4 q[2]; } ua;
        ua.q[0] = *(const uint4*)(aRow + kk + hi * 16);
        ua.q[1] = *(const uint4*)(aRow + kk + hi * 16 + 8);

        // B fragments: lanes 0-15: K = kk+0..15; lanes 16-31: K = kk+16..31
        v16bf Bv0 = loadB_bf16(wrow0 + kk + hi * 16);
        v16bf Bv1 = loadB_bf16(wrow1 + kk + hi * 16);

        acc0 = __builtin_amdgcn_wmma_f32_16x16x32_bf16(
                   false, ua.v, false, Bv0, (short)0, acc0, false, false);
        acc1 = __builtin_amdgcn_wmma_f32_16x16x32_bf16(
                   false, ua.v, false, Bv1, (short)0, acc1, false, false);
    }

    // ---- store: C/D layout — VGPR rr, lane l -> M = rr + 8*hi, N = l&15 ----
    const float bias0 = be[(size_t)e * CDIM + n0];
    const float bias1 = be[(size_t)e * CDIM + n1];
    if (m0 + 16 <= count) {
        // full tile: unguarded scatter
#pragma unroll
        for (int rr = 0; rr < 8; ++rr) {
            const int tok = sTok[rr + hi * 8];
            float* orow = out + (size_t)tok * CDIM;
            orow[n0] = acc0[rr] + bias0;
            orow[n1] = acc1[rr] + bias1;
        }
    } else {
#pragma unroll
        for (int rr = 0; rr < 8; ++rr) {
            const int gm = m0 + rr + hi * 8;
            if (gm < count) {
                const int tok = sTok[rr + hi * 8];
                float* orow = out + (size_t)tok * CDIM;
                orow[n0] = acc0[rr] + bias0;
                orow[n1] = acc1[rr] + bias1;
            }
        }
    }
}

// ---------------- host launcher ----------------
extern "C" void kernel_launch(void* const* d_in, const int* in_sizes, int n_in,
                              void* d_out, int out_size, void* d_ws, size_t ws_size,
                              hipStream_t stream) {
    const float* x  = (const float*)d_in[0];   // [B,T,C]
    const float* Wr = (const float*)d_in[1];   // [E,C]
    const float* br = (const float*)d_in[2];   // [E]
    const float* We = (const float*)d_in[3];   // [E,C,C]
    const float* be = (const float*)d_in[4];   // [E,C]
    float* out = (float*)d_out;                // [B,T,C]

    int* counts    = (int*)d_ws;               // NEXP ints
    int* tokenList = (int*)((char*)d_ws + 256);// NEXP * NTOK ints (~256 KB)

    moe_init_kernel<<<1, 32, 0, stream>>>(counts);

    moe_router_kernel<<<NTOK / 8, 256, 0, stream>>>(x, Wr, br, out, counts, tokenList);

    dim3 grid(NTOK / 16, CDIM / 256, NEXP);
    moe_expert_gemm_kernel<<<grid, 256, 0, stream>>>(x, We, be, out, counts, tokenList);
}
